// Temporal_aware_Neighbor_InteractionEncoder_41738492182952
// MI455X (gfx1250) — matrix operand order repaired and checked by reading, not compiled, not run on hardware
//
#include <hip/hip_runtime.h>
#include <hip/hip_bf16.h>

typedef __attribute__((ext_vector_type(16))) _Float16 v16h;
typedef __attribute__((ext_vector_type(8)))  float    v8f;

#define BB   1024
#define LL   256
#define DD   64
#define TDIM 100

__global__ __launch_bounds__(256)
void tani_fused_kernel(const int*   __restrict__ src_ids,
                       const int*   __restrict__ dst_ids,
                       const float* __restrict__ src_times,
                       const float* __restrict__ dst_times,
                       const float* __restrict__ node_times,
                       const float* __restrict__ w_time,
                       const float* __restrict__ b_time,
                       const float* __restrict__ w_ts,
                       const float* __restrict__ b_ts,
                       const float* __restrict__ w1,
                       const float* __restrict__ b1,
                       const float* __restrict__ w2,
                       const float* __restrict__ b2,
                       float*       __restrict__ out)
{
    __shared__ int      s_src[LL];
    __shared__ int      s_dst[LL];
    __shared__ _Float16 s_h[LL * DD];      // 32 KB, reused for src then dst side
    __shared__ _Float16 s_w2t[DD * DD];    // w2 transposed: [n][k]
    __shared__ float    s_wtime[TDIM], s_btime[TDIM], s_wts[TDIM];
    __shared__ float    s_w1[DD], s_b1[DD], s_b2[DD];

    const int b = blockIdx.x;     // row
    const int t = threadIdx.x;    // 0..255 == l index

    // ---- stage ids and small parameter vectors into LDS ----
    s_src[t] = src_ids[b * LL + t];
    s_dst[t] = dst_ids[b * LL + t];
    if (t < TDIM) { s_wtime[t] = w_time[t]; s_btime[t] = b_time[t]; s_wts[t] = w_ts[t]; }
    if (t < DD)   { s_w1[t] = w1[t]; s_b1[t] = b1[t]; s_b2[t] = b2[t]; }
    #pragma unroll
    for (int i = t; i < DD * DD; i += 256) {
        int k = i >> 6, n = i & 63;
        s_w2t[n * DD + k] = (_Float16)w2[i];   // transpose so B frags are contiguous in K
    }
    __syncthreads();

    // ---- histogram-gather collapsed to row-local equality counts ----
    const int my_s = s_src[t];
    const int my_d = s_dst[t];
    int css = 0, csd = 0, cds = 0, cdd = 0;
    for (int j = 0; j < LL; ++j) {
        int vs = s_src[j], vd = s_dst[j];   // LDS broadcast reads
        css += (vs == my_s);
        csd += (vd == my_s);
        cds += (vs == my_d);
        cdd += (vd == my_d);
    }
    if (my_s == 0) { css = 0; csd = 0; }    // id 0 is padding -> histogram entry stays 0
    if (my_d == 0) { cds = 0; cdd = 0; }

    // ---- LTF time encoder -> scalar -> sigmoid (fast transcendentals) ----
    const float ntm  = node_times[b];
    const float td_s = ntm - src_times[b * LL + t];
    const float td_d = ntm - dst_times[b * LL + t];
    float acc_s = 0.f, acc_d = 0.f;
    for (int k = 0; k < TDIM; ++k) {
        float w  = s_wtime[k], bo = s_btime[k], ws = s_wts[k];
        acc_s += __cosf(fmaf(td_s, w, bo)) * ws;
        acc_d += __cosf(fmaf(td_d, w, bo)) * ws;
    }
    const float bts = b_ts[0];
    const float sw  = 1.f / (1.f + __expf(-(acc_s + bts)));
    const float dw  = 1.f / (1.f + __expf(-(acc_d + bts)));

    const float wc[2][2] = { { css * sw, csd * sw },     // src side
                             { cds * dw, cdd * dw } };   // dst side

    const int wave  = t >> 5;
    const int lane  = t & 31;
    const int lrow  = lane & 15;
    const int lhalf = lane >> 4;            // 0: lanes 0-15, 1: lanes 16-31
    const int kofsA = lhalf << 3;           // A-frag K chunk offset (0 or 8)
    const int kofsB = lhalf << 3;           // B-frag K chunk offset (0 or 8)

    for (int side = 0; side < 2; ++side) {
        __syncthreads();   // previous GEMM pass finished reading s_h

        // hsum[l][d] = relu(c0*w1+b1) + relu(c1*w1+b1)  (matmul linearity folds the "2" axis)
        const float c0 = wc[side][0], c1 = wc[side][1];
        #pragma unroll
        for (int d = 0; d < DD; ++d) {
            float h0 = fmaxf(fmaf(c0, s_w1[d], s_b1[d]), 0.f);
            float h1 = fmaxf(fmaf(c1, s_w1[d], s_b1[d]), 0.f);
            s_h[t * DD + d] = (_Float16)(h0 + h1);
        }
        __syncthreads();

        // ---- per-row GEMM: (256 x 64) @ (64 x 64) via v_wmma_f32_16x16x32_f16 ----
        float* op = out + (size_t)side * (size_t)BB * LL * DD + (size_t)b * LL * DD;
        for (int job = wave; job < 64; job += 8) {        // 16 M-tiles x 4 N-tiles
            const int m0 = (job >> 2) << 4;
            const int n0 = (job & 3) << 4;
            v8f acc = {};
            #pragma unroll
            for (int k0 = 0; k0 < DD; k0 += 32) {
                // A fragment: row = m0+lrow; K runs [base,base+8) and [base+16,base+24)
                const _Float16* ap = &s_h[(m0 + lrow) * DD + k0 + kofsA];
                // B fragment: col = n0+lrow; K runs [base,base+8) and [base+16,base+24) (w2 transposed)
                const _Float16* bp = &s_w2t[(n0 + lrow) * DD + k0 + kofsB];
                v16h av, bv;
                #pragma unroll
                for (int e = 0; e < 8; ++e) {
                    av[e]     = ap[e];
                    av[8 + e] = ap[16 + e];
                }
                #pragma unroll
                for (int e = 0; e < 8; ++e) { bv[e] = bp[e]; bv[8 + e] = bp[16 + e]; }
                acc = __builtin_amdgcn_wmma_f32_16x16x32_f16(
                          false, av, false, bv, (short)0, acc, false, false);
            }
            const int   n     = n0 + lrow;
            const float bias2 = 2.f * s_b2[n];            // sum over the "2" axis adds b2 twice
            const int   mbase = m0 + (lhalf << 3);        // D layout: rows r / 8+r
            #pragma unroll
            for (int r = 0; r < 8; ++r)
                op[(size_t)(mbase + r) * DD + n] = acc[r] + bias2;
        }
    }
}

extern "C" void kernel_launch(void* const* d_in, const int* in_sizes, int n_in,
                              void* d_out, int out_size, void* d_ws, size_t ws_size,
                              hipStream_t stream) {
    const int*   src_ids    = (const int*)  d_in[0];
    const int*   dst_ids    = (const int*)  d_in[1];
    const float* src_times  = (const float*)d_in[2];
    const float* dst_times  = (const float*)d_in[3];
    const float* node_times = (const float*)d_in[4];
    // d_in[5] = num_nodes (unused: histogram collapsed to row-local counts)
    const float* w_time     = (const float*)d_in[6];
    const float* b_time     = (const float*)d_in[7];
    const float* w_ts       = (const float*)d_in[8];
    const float* b_ts       = (const float*)d_in[9];
    const float* w1         = (const float*)d_in[10];
    const float* b1         = (const float*)d_in[11];
    const float* w2         = (const float*)d_in[12];
    const float* b2         = (const float*)d_in[13];
    float*       out        = (float*)d_out;

    tani_fused_kernel<<<dim3(BB), dim3(256), 0, stream>>>(
        src_ids, dst_ids, src_times, dst_times, node_times,
        w_time, b_time, w_ts, b_ts, w1, b1, w2, b2, out);
}
